// OverlapRowParallelLayer_12859132084237
// MI455X (gfx1250) — matrix operand order repaired and checked by reading, not compiled, not run on hardware
//
#include <hip/hip_runtime.h>

typedef _Float16 v8h  __attribute__((ext_vector_type(8)));
typedef _Float16 v16h __attribute__((ext_vector_type(16)));
typedef float    v8f  __attribute__((ext_vector_type(8)));

#define MDIM 2048
#define KDIM 4096
#define NDIM 8192

#define BM 128
#define BN 128
#define BK 32
#define KT (KDIM / BK)     // 128 K-steps
#define LDSROW 40          // halves per LDS row: 32 data + 8 pad (80B, 16B aligned)

// CDNA5 async global->LDS copy, 16B per lane, GV addressing (SADDR = off).
// vdst = per-lane LDS byte offset, vaddr = per-lane 64-bit global address.
// Tracked by ASYNCcnt; completion enforced with s_wait_asynccnt.
__device__ __forceinline__ void async_copy_b128(unsigned lds_byte_off, const void* gaddr) {
    asm volatile("global_load_async_to_lds_b128 %0, %1, off"
                 :: "v"(lds_byte_off), "v"(gaddr)
                 : "memory");
}

__device__ __forceinline__ void wait_async0() {
    asm volatile("s_wait_asynccnt 0x0" ::: "memory");
}

// LDS byte offset of a generic (__shared__-derived) pointer: flat aperture maps
// addr[31:0] onto the LDS space, so truncation yields the DS-instruction offset.
__device__ __forceinline__ unsigned lds_off_of(const void* p) {
    return (unsigned)(uintptr_t)p;
}

// out = x @ weight^T  (fp16 inputs, f32 accumulate)
// Block: 256 threads (8 wave32), tile 128x128x32, wave tile 32x64 (2x4 WMMA)
__global__ __launch_bounds__(256)
void gemm_f16_wmma(const _Float16* __restrict__ X,
                   const _Float16* __restrict__ W,
                   float* __restrict__ Out)
{
    __shared__ _Float16 sA[2][BM * LDSROW];
    __shared__ _Float16 sB[2][BN * LDSROW];

    const int tid  = threadIdx.x;
    const int lane = tid & 31;
    const int wave = tid >> 5;
    const int wm = wave & 3;        // 4 waves along M -> 4*32 = 128
    const int wn = wave >> 2;       // 2 waves along N -> 2*64 = 128
    const int laneRow = lane & 15;  // row within a 16x16 tile
    const int laneHi  = lane >> 4;  // K-halves group selector

    const int bm = blockIdx.y * BM;
    const int bn = blockIdx.x * BN;

    // ---- global -> LDS staging: each tile is 128 rows x 32 halves = 512 x 16B chunks,
    //      2 chunks per thread (rows t>>2 and t>>2 + 64, chunk (t&3) within the row)
    const int rowLd  = tid >> 2;          // 0..63
    const int chunkH = (tid & 3) * 8;     // half-offset within row (16B chunks)

    const _Float16* gA = X + (size_t)(bm + rowLd) * KDIM + chunkH;
    const _Float16* gB = W + (size_t)(bn + rowLd) * KDIM + chunkH;

    const int ldsOff0 = rowLd * LDSROW + chunkH;           // bytes: rowLd*80 + 16*(t&3) -> 16B aligned
    const int ldsOff1 = (rowLd + 64) * LDSROW + chunkH;

    // Per-buffer LDS byte offsets for this thread's 4 staging chunks.
    unsigned dA0[2], dA1[2], dB0[2], dB1[2];
#pragma unroll
    for (int b = 0; b < 2; ++b) {
        dA0[b] = lds_off_of(&sA[b][ldsOff0]);
        dA1[b] = lds_off_of(&sA[b][ldsOff1]);
        dB0[b] = lds_off_of(&sB[b][ldsOff0]);
        dB1[b] = lds_off_of(&sB[b][ldsOff1]);
    }

    // preload K-tile 0 straight into LDS (no VGPR staging)
    async_copy_b128(dA0[0], gA);
    async_copy_b128(dA1[0], gA + (size_t)64 * KDIM);
    async_copy_b128(dB0[0], gB);
    async_copy_b128(dB1[0], gB + (size_t)64 * KDIM);
    wait_async0();
    __syncthreads();

    v8f acc[2][4];
#pragma unroll
    for (int mi = 0; mi < 2; ++mi)
#pragma unroll
        for (int ni = 0; ni < 4; ++ni)
            acc[mi][ni] = (v8f){0.f,0.f,0.f,0.f,0.f,0.f,0.f,0.f};

    // Fragment base addresses in LDS (constant across K loop).
    // A 16x32 f16 layout (ISA 7.12.2): lanes 0-15 hold K 0-7 & 16-23; lanes 16-31 hold K 8-15 & 24-31.
    // B 32x16 f16 layout: lanes 0-15 hold K 0-7 & 16-23 of column n; lanes 16-31 hold K 8-15 & 24-31.
    int aBase[2], bBase[4];
#pragma unroll
    for (int mi = 0; mi < 2; ++mi)
        aBase[mi] = (wm * 32 + mi * 16 + laneRow) * LDSROW + laneHi * 8;
#pragma unroll
    for (int ni = 0; ni < 4; ++ni)
        bBase[ni] = (wn * 64 + ni * 16 + laneRow) * LDSROW + laneHi * 8;

    for (int kt = 0; kt < KT; ++kt) {
        const int cur = kt & 1;
        const bool hasNext = (kt + 1) < KT;

        // L2 prefetch a few tiles ahead (global_prefetch_b8)
        if (kt + 4 < KT) {
            __builtin_prefetch(gA + (size_t)(kt + 4) * BK, 0, 0);
            __builtin_prefetch(gB + (size_t)(kt + 4) * BK, 0, 0);
        }

        // kick off async copies of K-tile kt+1 directly into the other LDS buffer;
        // the barrier that ended iteration kt-1 guarantees nobody still reads it
        if (hasNext) {
            const int nxt = cur ^ 1;
            const _Float16* pa = gA + (size_t)(kt + 1) * BK;
            const _Float16* pb = gB + (size_t)(kt + 1) * BK;
            async_copy_b128(dA0[nxt], pa);
            async_copy_b128(dA1[nxt], pa + (size_t)64 * KDIM);
            async_copy_b128(dB0[nxt], pb);
            async_copy_b128(dB1[nxt], pb + (size_t)64 * KDIM);
        }

        // load fragments from LDS (two 16B ds loads per fragment, 16B aligned)
        v16h aF[2], bF[4];
#pragma unroll
        for (int mi = 0; mi < 2; ++mi) {
            v8h lo = *(const v8h*)(&sA[cur][aBase[mi]]);
            v8h hi = *(const v8h*)(&sA[cur][aBase[mi] + 16]);
            aF[mi] = __builtin_shufflevector(lo, hi, 0,1,2,3,4,5,6,7,8,9,10,11,12,13,14,15);
        }
#pragma unroll
        for (int ni = 0; ni < 4; ++ni) {
            v8h lo = *(const v8h*)(&sB[cur][bBase[ni]]);
            v8h hi = *(const v8h*)(&sB[cur][bBase[ni] + 16]);
            bF[ni] = __builtin_shufflevector(lo, hi, 0,1,2,3,4,5,6,7,8,9,10,11,12,13,14,15);
        }

        // 8 x v_wmma_f32_16x16x32_f16 per wave per K-step (async copy latency hides behind these)
#pragma unroll
        for (int mi = 0; mi < 2; ++mi)
#pragma unroll
            for (int ni = 0; ni < 4; ++ni)
                acc[mi][ni] = __builtin_amdgcn_wmma_f32_16x16x32_f16(
                    /*neg_a=*/false, aF[mi], /*neg_b=*/false, bF[ni],
                    /*c_mod=*/(short)0, acc[mi][ni],
                    /*reuse_a=*/false, /*reuse_b=*/false);

        // publish next tile: drain ASYNCcnt (barrier does not cover it), then barrier
        wait_async0();
        __syncthreads();
    }

    // Epilogue: C/D layout — lane l holds col (l&15), rows (l>>4)*8 + r in VGPR r.
    // Lanes 0-15 write consecutive columns -> coalesced f32 stores.
#pragma unroll
    for (int mi = 0; mi < 2; ++mi) {
#pragma unroll
        for (int ni = 0; ni < 4; ++ni) {
            const int col  = bn + wn * 64 + ni * 16 + laneRow;
            const int row0 = bm + wm * 32 + mi * 16 + laneHi * 8;
#pragma unroll
            for (int r = 0; r < 8; ++r) {
                Out[(size_t)(row0 + r) * NDIM + col] = acc[mi][ni][r];
            }
        }
    }
}

extern "C" void kernel_launch(void* const* d_in, const int* in_sizes, int n_in,
                              void* d_out, int out_size, void* d_ws, size_t ws_size,
                              hipStream_t stream) {
    const _Float16* x = (const _Float16*)d_in[0];   // [M,K] fp16 row-major
    const _Float16* w = (const _Float16*)d_in[1];   // [N,K] fp16 row-major
    float* out = (float*)d_out;                     // [M,N]

    dim3 grid(NDIM / BN, MDIM / BM);                // (64, 16) = 1024 workgroups
    dim3 block(256);                                // 8 wave32
    gemm_f16_wmma<<<grid, block, 0, stream>>>(x, w, out);
}
